// SPINN_24464133718536
// MI455X (gfx1250) — matrix operand order repaired and checked by reading, not compiled, not run on hardware
//
#include <hip/hip_runtime.h>
#include <math.h>

// Problem constants (match reference)
#define Bn   128
#define Ln   64
#define Dn   1024
#define DTn  256
#define Tn   127        // 2L-1
#define Sn   64         // max stack depth
#define KTR  3328       // 3*D + DT   (tracker GEMM K)
#define KRD  2304       // 2*D + DT   (reduce GEMM K)
#define NTR  1024       // 4*DT
#define NRD  5120       // 5*D
#define KT_TR (KTR/32)  // 104 k-tiles
#define KT_RD (KRD/32)  // 72 k-tiles
#define KSPLIT 4        // split-K factor; chunks 26 / 18 (both even)

typedef __attribute__((ext_vector_type(16))) __bf16 v16bf;
typedef __attribute__((ext_vector_type(8)))  __bf16 v8bf;
typedef __attribute__((ext_vector_type(8)))  float  v8f;

union V16U { v16bf v; v8bf h[2]; };

__device__ __forceinline__ float sigm(float x) { return 1.0f / (1.0f + expf(-x)); }
__device__ __forceinline__ int iclamp(int x, int lo, int hi) {
  return x < lo ? lo : (x > hi ? hi : x);
}

// ---------------------------------------------------------------------------
// Weight packing: fp32 row-major [K, N] concat -> bf16 fragments.
// Fragment (nt, kt) is 512 contiguous bf16 at ((nt*Kt + kt)*512).
// Within a fragment, element (lane, j):  n = nt*16 + (lane&15),
//                                        k = kt*32 + (lane>>4)*16 + j
// (Contiguous k order used here, same as A: lanes 0-15 hold K=0..15, lanes 16-31 hold K=16..31.)
// ---------------------------------------------------------------------------
__global__ void __launch_bounds__(256)
pack_tracker_w(const float* __restrict__ Wb, const float* __restrict__ Ws1,
               const float* __restrict__ Ws2, const float* __restrict__ Wlat,
               __bf16* __restrict__ out) {
  int idx = blockIdx.x * blockDim.x + threadIdx.x;
  const int total = KT_TR * (NTR / 16) * 512;
  if (idx >= total) return;
  int frag = idx >> 9, r = idx & 511;
  int lane = r >> 4, j = r & 15;
  int nt = frag / KT_TR, kt = frag % KT_TR;
  int n = nt * 16 + (lane & 15);
  int k = kt * 32 + (lane >> 4) * 16 + j;
  float w;
  if      (k < 1024) w = Wb  [(size_t)k          * NTR + n];
  else if (k < 2048) w = Ws1 [(size_t)(k - 1024) * NTR + n];
  else if (k < 3072) w = Ws2 [(size_t)(k - 2048) * NTR + n];
  else               w = Wlat[(size_t)(k - 3072) * NTR + n];
  out[idx] = (__bf16)w;
}

__global__ void __launch_bounds__(256)
pack_reduce_w(const float* __restrict__ Wrl, const float* __restrict__ Wrr,
              const float* __restrict__ Wrt, __bf16* __restrict__ out) {
  int idx = blockIdx.x * blockDim.x + threadIdx.x;
  const int total = KT_RD * (NRD / 16) * 512;
  if (idx >= total) return;
  int frag = idx >> 9, r = idx & 511;
  int lane = r >> 4, j = r & 15;
  int nt = frag / KT_RD, kt = frag % KT_RD;
  int n = nt * 16 + (lane & 15);
  int k = kt * 32 + (lane >> 4) * 16 + j;
  float w;
  if      (k < 1024) w = Wrl[(size_t)k          * NRD + n];
  else if (k < 2048) w = Wrr[(size_t)(k - 1024) * NRD + n];
  else               w = Wrt[(size_t)(k - 2048) * NRD + n];
  out[idx] = (__bf16)w;
}

// ---------------------------------------------------------------------------
// Split-K bf16 WMMA GEMM:
//   Cpart[part][128, N] = A[128, ktChunk*32 slice] @ Wpacked slice
// One wave -> 16x64 C strip (4 fragments), double-buffered k-tile pipeline.
// blockDim=256 (8 waves = 8 row tiles); grid = KSPLIT * (N/64) blocks:
//   ns = blockIdx % (N/64), part = blockIdx / (N/64).
// ---------------------------------------------------------------------------
struct Frags {
  V16U a;
  V16U b[4];
};

__device__ __forceinline__ void load_frags(Frags& fr, const __bf16* __restrict__ Arow,
                                           const __bf16* __restrict__ Bbase,
                                           size_t bstride, int kt) {
  fr.a.h[0] = *(const v8bf*)(Arow + kt * 32);
  fr.a.h[1] = *(const v8bf*)(Arow + kt * 32 + 8);
#pragma unroll
  for (int f = 0; f < 4; ++f) {
    const __bf16* bp = Bbase + (size_t)f * bstride + (size_t)kt * 512;
    fr.b[f].h[0] = *(const v8bf*)(bp);
    fr.b[f].h[1] = *(const v8bf*)(bp + 8);
  }
}

__global__ void __launch_bounds__(256)
gemm_bf16_wmma_splitk(const __bf16* __restrict__ A, int K,
                      const __bf16* __restrict__ Wp, int Kt,
                      float* __restrict__ Cpart, int N, int ktChunk) {
  const int nstrips = N >> 6;
  const int part  = blockIdx.x / nstrips;
  const int ns    = blockIdx.x % nstrips;
  const int mt    = threadIdx.x >> 5;   // 8 row tiles over M=128
  const int lane  = threadIdx.x & 31;
  const int n0    = ns * 64;
  const int khalf = lane >> 4;
  const int kt0   = part * ktChunk;

  v8f acc[4];
#pragma unroll
  for (int f = 0; f < 4; ++f) acc[f] = (v8f){0.f,0.f,0.f,0.f,0.f,0.f,0.f,0.f};

  const int row = mt * 16 + (lane & 15);
  // A layout (matches packed B): lane m: K = khalf*16..+15 contiguous
  const __bf16* Arow  = A + (size_t)row * K + khalf * 16 + (size_t)kt0 * 32;
  const size_t  bstr  = (size_t)Kt * 512;   // elements between adjacent n-tiles
  const __bf16* Bbase = Wp + (size_t)(n0 >> 4) * bstr + lane * 16 + (size_t)kt0 * 512;

  Frags cur, nxt;
  load_frags(cur, Arow, Bbase, bstr, 0);
  // ktChunk is even (26 / 18): unroll-by-2 ping-pong
  for (int kt = 0; kt < ktChunk; kt += 2) {
    load_frags(nxt, Arow, Bbase, bstr, kt + 1);
#pragma unroll
    for (int f = 0; f < 4; ++f)
      acc[f] = __builtin_amdgcn_wmma_f32_16x16x32_bf16(
          false, cur.a.v, false, cur.b[f].v, (short)0, acc[f], false, false);
    if (kt + 2 < ktChunk) load_frags(cur, Arow, Bbase, bstr, kt + 2);
#pragma unroll
    for (int f = 0; f < 4; ++f)
      acc[f] = __builtin_amdgcn_wmma_f32_16x16x32_bf16(
          false, nxt.a.v, false, nxt.b[f].v, (short)0, acc[f], false, false);
  }

  // C layout: VGPR r -> M = r (lanes 0-15) / r+8 (lanes 16-31); N = lane&15
  float* Cp = Cpart + (size_t)part * Bn * N;
  const int rbase = mt * 16 + khalf * 8;
  const int cb = n0 + (lane & 15);
#pragma unroll
  for (int f = 0; f < 4; ++f)
#pragma unroll
    for (int r = 0; r < 8; ++r)
      Cp[(size_t)(rbase + r) * N + cb + f * 16] = acc[f][r];
}

// ---------------------------------------------------------------------------
// Per-step kernels
// ---------------------------------------------------------------------------
__global__ void __launch_bounds__(256)
init_state(float* th, float* tc, int* ptr, int* bptr) {
  int idx = blockIdx.x * blockDim.x + threadIdx.x;
  if (idx < Bn * DTn) { th[idx] = 0.f; tc[idx] = 0.f; }
  if (idx < Bn)       { ptr[idx] = 0;  bptr[idx] = 0; }
}

__global__ void __launch_bounds__(256)
gather_step(const float* __restrict__ bufH, const float* __restrict__ bufC,
            const float* __restrict__ stackH, const float* __restrict__ stackC,
            const int* __restrict__ ptr, const int* __restrict__ bptr,
            const float* __restrict__ th,
            float* __restrict__ bh_g, float* __restrict__ bc_g,
            float* __restrict__ s1c_g, float* __restrict__ s2c_g,
            __bf16* __restrict__ Atr, __bf16* __restrict__ Ared) {
  int idx = blockIdx.x * blockDim.x + threadIdx.x;
  if (idx >= Bn * Dn) return;
  int b = idx >> 10, d = idx & 1023;
  int p  = ptr[b];
  int bs = iclamp(bptr[b], 0, Ln - 1);
  int i1 = iclamp(p - 1, 0, Sn - 1);
  int i2 = iclamp(p - 2, 0, Sn - 1);
  float bh = bufH[((size_t)b * Ln + bs) * Dn + d];
  float bc = bufC[((size_t)b * Ln + bs) * Dn + d];
  float s1h = 0.f, s1c = 0.f, s2h = 0.f, s2c = 0.f;
  if (p > 0) { s1h = stackH[((size_t)b * Sn + i1) * Dn + d];
               s1c = stackC[((size_t)b * Sn + i1) * Dn + d]; }
  if (p > 1) { s2h = stackH[((size_t)b * Sn + i2) * Dn + d];
               s2c = stackC[((size_t)b * Sn + i2) * Dn + d]; }
  bh_g[idx] = bh; bc_g[idx] = bc; s1c_g[idx] = s1c; s2c_g[idx] = s2c;
  __bf16* at = Atr + (size_t)b * KTR;
  at[d]        = (__bf16)bh;
  at[1024 + d] = (__bf16)s1h;
  at[2048 + d] = (__bf16)s2h;
  if (d < DTn) at[3072 + d] = (__bf16)th[b * DTn + d];
  __bf16* ar = Ared + (size_t)b * KRD;
  ar[d]        = (__bf16)s2h;
  ar[1024 + d] = (__bf16)s1h;
}

__global__ void __launch_bounds__(256)
tracker_pw(const float* __restrict__ gatesP, const float* __restrict__ blat,
           float* __restrict__ th, float* __restrict__ tc,
           __bf16* __restrict__ Ared, int tstep) {
  int idx = blockIdx.x * blockDim.x + threadIdx.x;
  if (idx >= Bn * DTn) return;
  int b = idx >> 8, j = idx & 255;
  float a = 0.f, i = 0.f, f = 0.f, o = 0.f;
#pragma unroll
  for (int p4 = 0; p4 < KSPLIT; ++p4) {
    const float* g = gatesP + ((size_t)p4 * Bn + b) * NTR;
    a += g[j]; i += g[256 + j]; f += g[512 + j]; o += g[768 + j];
  }
  if (tstep > 0) {
    a += blat[j]; i += blat[256 + j]; f += blat[512 + j]; o += blat[768 + j];
  }
  float c = tanhf(a) * sigm(i) + sigm(f) * tc[idx];
  float h = sigm(o) * tanhf(c);
  tc[idx] = c; th[idx] = h;
  Ared[(size_t)b * KRD + 2048 + j] = (__bf16)h;
}

__global__ void __launch_bounds__(256)
reduce_pw(const float* __restrict__ rgP, const float* __restrict__ brl,
          const float* __restrict__ bh_g, const float* __restrict__ bc_g,
          const float* __restrict__ s1c_g, const float* __restrict__ s2c_g,
          const int* __restrict__ ptr, const int* __restrict__ trans, int tstep,
          float* __restrict__ stackH, float* __restrict__ stackC) {
  int idx = blockIdx.x * blockDim.x + threadIdx.x;
  if (idx >= Bn * Dn) return;
  int b = idx >> 10, d = idx & 1023;
  int tr = trans[b * Tn + tstep];
  if (tr != 0 && tr != 1) return;  // SKIP
  float nh, nc;
  if (tr == 1) {
    float a = 0.f, ig = 0.f, f1 = 0.f, f2 = 0.f, o = 0.f;
#pragma unroll
    for (int p4 = 0; p4 < KSPLIT; ++p4) {
      const float* g = rgP + ((size_t)p4 * Bn + b) * NRD;
      a  += g[d];
      ig += g[1024 + d];
      f1 += g[2048 + d];
      f2 += g[3072 + d];
      o  += g[4096 + d];
    }
    a += brl[d]; ig += brl[1024 + d]; f1 += brl[2048 + d];
    f2 += brl[3072 + d]; o += brl[4096 + d];
    float rc = tanhf(a) * sigm(ig) + sigm(f1) * s2c_g[idx] + sigm(f2) * s1c_g[idx];
    nh = sigm(o) * tanhf(rc);
    nc = rc;
  } else {
    nh = bh_g[idx];
    nc = bc_g[idx];
  }
  int p = ptr[b];
  int pos = (tr == 0) ? iclamp(p, 0, Sn - 1) : iclamp(p - 2, 0, Sn - 1);
  stackH[((size_t)b * Sn + pos) * Dn + d] = nh;
  stackC[((size_t)b * Sn + pos) * Dn + d] = nc;
}

__global__ void update_ptrs(int* ptr, int* bptr, const int* __restrict__ trans, int tstep) {
  int b = threadIdx.x;
  if (b >= Bn) return;
  int tr = trans[b * Tn + tstep];
  if (tr == 0)      { ptr[b] += 1; bptr[b] += 1; }
  else if (tr == 1) { ptr[b] -= 1; }
}

__global__ void __launch_bounds__(256)
finalize(const float* __restrict__ stackH, const float* __restrict__ stackC,
         const int* __restrict__ ptr, float* __restrict__ out) {
  int idx = blockIdx.x * blockDim.x + threadIdx.x;
  if (idx >= Bn * Dn) return;
  int b = idx >> 10, d = idx & 1023;
  int top = iclamp(ptr[b] - 1, 0, Sn - 1);
  out[(size_t)b * 2048 + d]        = stackH[((size_t)b * Sn + top) * Dn + d];
  out[(size_t)b * 2048 + 1024 + d] = stackC[((size_t)b * Sn + top) * Dn + d];
}

// ---------------------------------------------------------------------------
extern "C" void kernel_launch(void* const* d_in, const int* in_sizes, int n_in,
                              void* d_out, int out_size, void* d_ws, size_t ws_size,
                              hipStream_t stream) {
  (void)in_sizes; (void)n_in; (void)out_size; (void)ws_size;
  const float* bufH  = (const float*)d_in[0];
  const float* bufC  = (const float*)d_in[1];
  const float* Wb    = (const float*)d_in[2];
  const float* Ws1   = (const float*)d_in[3];
  const float* Ws2   = (const float*)d_in[4];
  const float* Wlat  = (const float*)d_in[5];
  const float* blat  = (const float*)d_in[6];
  const float* Wrl   = (const float*)d_in[7];
  const float* Wrr   = (const float*)d_in[8];
  const float* Wrt   = (const float*)d_in[9];
  const float* brl   = (const float*)d_in[10];
  const int*   trans = (const int*)d_in[11];
  float* out = (float*)d_out;

  char* ws = (char*)d_ws;
  size_t off = 0;
  auto carve = [&](size_t bytes) -> char* {
    char* p = ws + off;
    off = (off + bytes + 255) & ~(size_t)255;
    return p;
  };
  float*  stackH = (float*) carve((size_t)Bn * Sn * Dn * 4);
  float*  stackC = (float*) carve((size_t)Bn * Sn * Dn * 4);
  __bf16* WtrP   = (__bf16*)carve((size_t)KT_TR * (NTR / 16) * 512 * 2);
  __bf16* WredP  = (__bf16*)carve((size_t)KT_RD * (NRD / 16) * 512 * 2);
  __bf16* Atr    = (__bf16*)carve((size_t)Bn * KTR * 2);
  __bf16* Ared   = (__bf16*)carve((size_t)Bn * KRD * 2);
  float*  gatesP = (float*) carve((size_t)KSPLIT * Bn * NTR * 4);
  float*  rgP    = (float*) carve((size_t)KSPLIT * Bn * NRD * 4);
  float*  bh_g   = (float*) carve((size_t)Bn * Dn * 4);
  float*  bc_g   = (float*) carve((size_t)Bn * Dn * 4);
  float*  s1c_g  = (float*) carve((size_t)Bn * Dn * 4);
  float*  s2c_g  = (float*) carve((size_t)Bn * Dn * 4);
  float*  th     = (float*) carve((size_t)Bn * DTn * 4);
  float*  tc     = (float*) carve((size_t)Bn * DTn * 4);
  int*    dptr   = (int*)   carve((size_t)Bn * 4);
  int*    dbptr  = (int*)   carve((size_t)Bn * 4);

  // One-time prep: bf16 weight packing + state init
  {
    int tot1 = KT_TR * (NTR / 16) * 512;
    int tot2 = KT_RD * (NRD / 16) * 512;
    pack_tracker_w<<<(tot1 + 255) / 256, 256, 0, stream>>>(Wb, Ws1, Ws2, Wlat, WtrP);
    pack_reduce_w<<<(tot2 + 255) / 256, 256, 0, stream>>>(Wrl, Wrr, Wrt, WredP);
    init_state<<<(Bn * DTn + 255) / 256, 256, 0, stream>>>(th, tc, dptr, dbptr);
  }

  const int gBD = (Bn * Dn + 255) / 256;    // 512 blocks
  for (int t = 0; t < Tn; ++t) {
    gather_step<<<gBD, 256, 0, stream>>>(bufH, bufC, stackH, stackC, dptr, dbptr,
                                         th, bh_g, bc_g, s1c_g, s2c_g, Atr, Ared);
    // tracker: KSPLIT * 16 strips * 8 row-tiles = 512 waves, 26 k-tiles each
    gemm_bf16_wmma_splitk<<<KSPLIT * (NTR / 64), 256, 0, stream>>>(
        Atr, KTR, WtrP, KT_TR, gatesP, NTR, KT_TR / KSPLIT);
    tracker_pw<<<(Bn * DTn + 255) / 256, 256, 0, stream>>>(gatesP, blat, th, tc, Ared, t);
    // reduce: KSPLIT * 80 strips * 8 row-tiles = 2560 waves, 18 k-tiles each
    gemm_bf16_wmma_splitk<<<KSPLIT * (NRD / 64), 256, 0, stream>>>(
        Ared, KRD, WredP, KT_RD, rgP, NRD, KT_RD / KSPLIT);
    reduce_pw<<<gBD, 256, 0, stream>>>(rgP, brl, bh_g, bc_g, s1c_g, s2c_g,
                                       dptr, trans, t, stackH, stackC);
    update_ptrs<<<1, 128, 0, stream>>>(dptr, dbptr, trans, t);
  }
  finalize<<<gBD, 256, 0, stream>>>(stackH, stackC, dptr, out);
}
